// MultiheadAttentionEncoder_45148696216955
// MI455X (gfx1250) — matrix operand order repaired and checked by reading, not compiled, not run on hardware
//
#include <hip/hip_runtime.h>
#include <hip/hip_bf16.h>

// ---------------------------------------------------------------------------
// Types for CDNA5 WMMA (wave32, 16x16x32 bf16 -> f32 accumulate)
// ---------------------------------------------------------------------------
typedef __bf16 bf16_t;
typedef __attribute__((ext_vector_type(16))) __bf16 v16bf;
typedef __attribute__((ext_vector_type(8)))  __bf16 v8bf;
typedef __attribute__((ext_vector_type(8)))  float  v8f;

__device__ __forceinline__ v8f wmma_bf16(v16bf a, v16bf b, v8f c) {
  // args: (neg_a, A, neg_b, B, c_mod, C, reuse_a, reuse_b)
  return __builtin_amdgcn_wmma_f32_16x16x32_bf16(false, a, false, b,
                                                 (short)0, c, false, false);
}

// A fragment: 16x32 (MxK) bf16, row-major source with leading dim ldk.
// ISA layout: row = lane%16; halves 0..7 = K kb..kb+7, halves 8..15 = K 16+kb..,
// kb = 8*(lane/16).
__device__ __forceinline__ v16bf load_a_frag(const bf16_t* __restrict__ base,
                                             int ldk, int lane) {
  const int row = lane & 15;
  const int kb  = (lane >> 4) * 8;
  const bf16_t* p = base + (size_t)row * ldk;
  v8bf lo = *(const v8bf*)(p + kb);
  v8bf hi = *(const v8bf*)(p + 16 + kb);
  v16bf a;
#pragma unroll
  for (int i = 0; i < 8; ++i) { a[i] = lo[i]; a[i + 8] = hi[i]; }
  return a;
}

// Same layout but reading a 16x32 tile staged in LDS (row stride 32 halves).
__device__ __forceinline__ v16bf load_a_frag_lds(const bf16_t* p, int lane) {
  const int row = lane & 15;
  const int kb  = (lane >> 4) * 8;
  const bf16_t* q = p + row * 32;
  v8bf lo = *(const v8bf*)(q + kb);
  v8bf hi = *(const v8bf*)(q + 16 + kb);
  v16bf a;
#pragma unroll
  for (int i = 0; i < 8; ++i) { a[i] = lo[i]; a[i + 8] = hi[i]; }
  return a;
}

// B fragment: 32x16 (KxN) bf16 where memory holds B^T row-major, i.e.
// source[n][k] with leading dim ldk and B[k][n] = source[n][k].
// ISA layout: col = lane%16; halves 0..7 = K kb..kb+7, halves 8..15 = K 16+kb..,
// kb = 8*(lane/16).
__device__ __forceinline__ v16bf load_b_frag(const bf16_t* __restrict__ base,
                                             size_t ldk, int lane) {
  const int col = lane & 15;
  const int kb  = (lane >> 4) * 8;
  const bf16_t* p = base + (size_t)col * ldk;
  v8bf lo = *(const v8bf*)(p + kb);
  v8bf hi = *(const v8bf*)(p + 16 + kb);
  v16bf b;
#pragma unroll
  for (int i = 0; i < 8; ++i) { b[i] = lo[i]; b[i + 8] = hi[i]; }
  return b;
}

// ---------------------------------------------------------------------------
// Problem constants
// ---------------------------------------------------------------------------
#define BB 4
#define SS 1024
#define EE 1024
#define HH 16
#define HDD 64
#define INV_TEMP 0.03125f   // 1/sqrt(1024)

// ---------------------------------------------------------------------------
// Kernel: f32 -> bf16 cast (grid-stride)
// ---------------------------------------------------------------------------
__global__ void cast_f32_to_bf16(const float* __restrict__ src,
                                 bf16_t* __restrict__ dst, size_t n) {
  size_t i = (size_t)blockIdx.x * blockDim.x + threadIdx.x;
  size_t stride = (size_t)gridDim.x * blockDim.x;
  for (; i < n; i += stride) dst[i] = (bf16_t)src[i];
}

// ---------------------------------------------------------------------------
// Kernel: QKV projection GEMM.  C[m,n] = sum_k x[m,k]*w[n,k] + bias[n]
// M=4096 (B*S), N=3072 (3E), K=1024.  Block tile 128x64, 8 waves of 32x32.
// Epilogue splits interleaved heads: n -> (h, which, d), scales Q by 1/TEMP,
// stores Q,K as [B,H,S,HD] bf16 and V transposed as [B,H,HD,S] bf16.
// ---------------------------------------------------------------------------
__global__ __launch_bounds__(256) void qkv_gemm_kernel(
    const bf16_t* __restrict__ xb,    // [4096,1024]
    const bf16_t* __restrict__ wb,    // [3072,1024]
    const float*  __restrict__ bias,  // [3072]
    bf16_t* __restrict__ Qb, bf16_t* __restrict__ Kb, bf16_t* __restrict__ Vtb)
{
  const int lane = threadIdx.x & 31;
  const int wid  = threadIdx.x >> 5;
  const int wm = wid & 3, wn = wid >> 2;
  const int m0 = blockIdx.x * 128 + wm * 32;
  const int n0 = blockIdx.y * 64 + wn * 32;
  const int K = EE;

  v8f acc[2][2] = {};
  for (int k0 = 0; k0 < K; k0 += 32) {
    v16bf a0 = load_a_frag(xb + (size_t)m0 * K + k0, K, lane);
    v16bf a1 = load_a_frag(xb + (size_t)(m0 + 16) * K + k0, K, lane);
    v16bf b0 = load_b_frag(wb + (size_t)n0 * K + k0, K, lane);
    v16bf b1 = load_b_frag(wb + (size_t)(n0 + 16) * K + k0, K, lane);
    acc[0][0] = wmma_bf16(a0, b0, acc[0][0]);
    acc[0][1] = wmma_bf16(a0, b1, acc[0][1]);
    acc[1][0] = wmma_bf16(a1, b0, acc[1][0]);
    acc[1][1] = wmma_bf16(a1, b1, acc[1][1]);
  }

  const int colL = lane & 15, half = lane >> 4;
#pragma unroll
  for (int im = 0; im < 2; ++im)
#pragma unroll
    for (int in = 0; in < 2; ++in)
#pragma unroll
      for (int r = 0; r < 8; ++r) {
        const int m = m0 + im * 16 + r + 8 * half;
        const int n = n0 + in * 16 + colL;
        float v = acc[im][in][r] + bias[n];
        const int b = m >> 10, s = m & (SS - 1);
        const int h = n / (3 * HDD);
        const int rem = n - h * (3 * HDD);
        const int which = rem >> 6, d = rem & (HDD - 1);
        const size_t bh = (size_t)b * HH + h;
        if (which == 0)
          Qb[(bh * SS + s) * HDD + d] = (bf16_t)(v * INV_TEMP);
        else if (which == 1)
          Kb[(bh * SS + s) * HDD + d] = (bf16_t)v;
        else
          Vtb[(bh * HDD + d) * SS + s] = (bf16_t)v;
      }
}

// ---------------------------------------------------------------------------
// Kernel: fused attention with online softmax (flash-style).
// Grid: (S/128, H, B); block 256 threads = 8 waves; each wave owns 16 q-rows.
// scores never leave registers; prev (256 MB) streams once from HBM.
// ---------------------------------------------------------------------------
__global__ __launch_bounds__(256) void attn_kernel(
    const bf16_t* __restrict__ Qb,    // [B,H,S,HD] pre-scaled by 1/TEMP
    const bf16_t* __restrict__ Kb,    // [B,H,S,HD]
    const bf16_t* __restrict__ Vtb,   // [B,H,HD,S]
    const float*  __restrict__ prev,  // [B,H,S,S]
    const unsigned char* __restrict__ mask, // [S,S] bool
    bf16_t* __restrict__ valsb)       // [B,S,E] bf16
{
  __shared__ bf16_t lds_p[8][16 * 32];

  const int lane = threadIdx.x & 31;
  const int wid  = threadIdx.x >> 5;
  const int b = blockIdx.z, h = blockIdx.y;
  const int q0 = blockIdx.x * 128 + wid * 16;
  const size_t bh = (size_t)b * HH + h;

  const bf16_t* Qh  = Qb + bh * SS * HDD;
  const bf16_t* Kh  = Kb + bh * SS * HDD;
  const bf16_t* Vth = Vtb + bh * HDD * SS;
  const float*  Ph  = prev + bh * SS * SS;

  const int colL = lane & 15, half = lane >> 4;

  // Q A-fragments for d=[0,32) and [32,64), reused across the whole j loop.
  const v16bf aq0 = load_a_frag(Qh + (size_t)q0 * HDD, HDD, lane);
  const v16bf aq1 = load_a_frag(Qh + (size_t)q0 * HDD + 32, HDD, lane);

  float mrow[8], lrow[8];
#pragma unroll
  for (int r = 0; r < 8; ++r) { mrow[r] = -1e30f; lrow[r] = 0.f; }
  v8f oacc[4] = {};

  for (int j0 = 0; j0 < SS; j0 += 32) {
    // ---- scores S[16, j0..j0+32) = Q*K^T (already scaled) ----
    v8f sc[2];
#pragma unroll
    for (int jn = 0; jn < 2; ++jn) {
      v8f a = {};
      v16bf bk0 = load_b_frag(Kh + (size_t)(j0 + jn * 16) * HDD, HDD, lane);
      v16bf bk1 = load_b_frag(Kh + (size_t)(j0 + jn * 16) * HDD + 32, HDD, lane);
      a = wmma_bf16(aq0, bk0, a);
      a = wmma_bf16(aq1, bk1, a);
      sc[jn] = a;
    }

    // ---- add prev, apply mask, track per-row tile max ----
    float s2[2][8];
    float tmax[8];
#pragma unroll
    for (int r = 0; r < 8; ++r) tmax[r] = -1e30f;
#pragma unroll
    for (int jn = 0; jn < 2; ++jn)
#pragma unroll
      for (int r = 0; r < 8; ++r) {
        const int qq = q0 + r + 8 * half;
        const int kk = j0 + jn * 16 + colL;
        float s = sc[jn][r] + Ph[(size_t)qq * SS + kk];
        if (mask[(size_t)qq * SS + kk]) s = -1e30f;
        s2[jn][r] = s;
        tmax[r] = fmaxf(tmax[r], s);
      }
    // row reduction within each half-wave (16 lanes hold one row)
#pragma unroll
    for (int r = 0; r < 8; ++r) {
      float v = tmax[r];
      v = fmaxf(v, __shfl_xor(v, 1, 32));
      v = fmaxf(v, __shfl_xor(v, 2, 32));
      v = fmaxf(v, __shfl_xor(v, 4, 32));
      v = fmaxf(v, __shfl_xor(v, 8, 32));
      tmax[r] = v;
    }

    float scalef[8];
#pragma unroll
    for (int r = 0; r < 8; ++r) {
      const float nm = fmaxf(mrow[r], tmax[r]);
      scalef[r] = __expf(mrow[r] - nm);
      mrow[r] = nm;
    }
    float psum[8];
#pragma unroll
    for (int r = 0; r < 8; ++r) psum[r] = 0.f;
#pragma unroll
    for (int jn = 0; jn < 2; ++jn)
#pragma unroll
      for (int r = 0; r < 8; ++r) {
        const float p = __expf(s2[jn][r] - mrow[r]);
        s2[jn][r] = p;
        psum[r] += p;
      }
#pragma unroll
    for (int r = 0; r < 8; ++r) {
      float v = psum[r];
      v += __shfl_xor(v, 1, 32);
      v += __shfl_xor(v, 2, 32);
      v += __shfl_xor(v, 4, 32);
      v += __shfl_xor(v, 8, 32);
      lrow[r] = lrow[r] * scalef[r] + v;
    }
#pragma unroll
    for (int f = 0; f < 4; ++f)
#pragma unroll
      for (int r = 0; r < 8; ++r) oacc[f][r] *= scalef[r];

    // ---- stage P (16x32) through LDS: C-layout f32 -> A-layout bf16 ----
    bf16_t* pw = lds_p[wid];
#pragma unroll
    for (int jn = 0; jn < 2; ++jn)
#pragma unroll
      for (int r = 0; r < 8; ++r)
        pw[(r + 8 * half) * 32 + jn * 16 + colL] = (bf16_t)s2[jn][r];
    const v16bf ap = load_a_frag_lds(pw, lane);

    // ---- oacc += P @ V  (V^T layout -> contiguous per-lane B loads) ----
#pragma unroll
    for (int f = 0; f < 4; ++f) {
      v16bf bv = load_b_frag(Vth + (size_t)(f * 16) * SS + j0, SS, lane);
      oacc[f] = wmma_bf16(ap, bv, oacc[f]);
    }
  }

  // ---- epilogue: normalize and store as [B,S,E] bf16 (heads concatenated)
#pragma unroll
  for (int f = 0; f < 4; ++f)
#pragma unroll
    for (int r = 0; r < 8; ++r) {
      const int qq = q0 + r + 8 * half;
      const int d  = f * 16 + colL;
      const float v = oacc[f][r] / lrow[r];
      valsb[((size_t)b * SS + qq) * EE + h * HDD + d] = (bf16_t)v;
    }
}

// ---------------------------------------------------------------------------
// Kernel: output projection.  out[m,n] = sum_k vals[m,k]*o_w[n,k] + o_b[n]
// M=4096, N=1024, K=1024.  Same tiling as the QKV GEMM; f32 output.
// ---------------------------------------------------------------------------
__global__ __launch_bounds__(256) void oproj_gemm_kernel(
    const bf16_t* __restrict__ A,     // [4096,1024] bf16
    const bf16_t* __restrict__ W,     // [1024,1024] bf16
    const float*  __restrict__ bias,  // [1024]
    float* __restrict__ out)          // [4096,1024] f32
{
  const int lane = threadIdx.x & 31;
  const int wid  = threadIdx.x >> 5;
  const int wm = wid & 3, wn = wid >> 2;
  const int m0 = blockIdx.x * 128 + wm * 32;
  const int n0 = blockIdx.y * 64 + wn * 32;
  const int K = EE;

  v8f acc[2][2] = {};
  for (int k0 = 0; k0 < K; k0 += 32) {
    v16bf a0 = load_a_frag(A + (size_t)m0 * K + k0, K, lane);
    v16bf a1 = load_a_frag(A + (size_t)(m0 + 16) * K + k0, K, lane);
    v16bf b0 = load_b_frag(W + (size_t)n0 * K + k0, K, lane);
    v16bf b1 = load_b_frag(W + (size_t)(n0 + 16) * K + k0, K, lane);
    acc[0][0] = wmma_bf16(a0, b0, acc[0][0]);
    acc[0][1] = wmma_bf16(a0, b1, acc[0][1]);
    acc[1][0] = wmma_bf16(a1, b0, acc[1][0]);
    acc[1][1] = wmma_bf16(a1, b1, acc[1][1]);
  }

  const int colL = lane & 15, half = lane >> 4;
#pragma unroll
  for (int im = 0; im < 2; ++im)
#pragma unroll
    for (int in = 0; in < 2; ++in)
#pragma unroll
      for (int r = 0; r < 8; ++r) {
        const int m = m0 + im * 16 + r + 8 * half;
        const int n = n0 + in * 16 + colL;
        out[(size_t)m * EE + n] = acc[im][in][r] + bias[n];
      }
}

// ---------------------------------------------------------------------------
// Host launcher
// ---------------------------------------------------------------------------
extern "C" void kernel_launch(void* const* d_in, const int* in_sizes, int n_in,
                              void* d_out, int out_size, void* d_ws, size_t ws_size,
                              hipStream_t stream) {
  (void)in_sizes; (void)n_in; (void)out_size; (void)ws_size;
  const float* x      = (const float*)d_in[0];
  const unsigned char* mask = (const unsigned char*)d_in[1]; // bool [1,1,S,S]
  const float* prev   = (const float*)d_in[2];
  const float* qkv_w  = (const float*)d_in[3];
  const float* qkv_b  = (const float*)d_in[4];
  const float* o_w    = (const float*)d_in[5];
  const float* o_b    = (const float*)d_in[6];
  float* out = (float*)d_out;

  char* ws = (char*)d_ws;
  size_t off = 0;
  auto walloc = [&](size_t bytes) -> void* {
    void* p = ws + off;
    off += (bytes + 255) & ~(size_t)255;
    return p;
  };
  const size_t M = (size_t)BB * SS;                 // 4096
  bf16_t* xb    = (bf16_t*)walloc(M * EE * sizeof(bf16_t));          // 8 MB
  bf16_t* wqkvb = (bf16_t*)walloc((size_t)3 * EE * EE * sizeof(bf16_t)); // 6 MB
  bf16_t* owb   = (bf16_t*)walloc((size_t)EE * EE * sizeof(bf16_t)); // 2 MB
  bf16_t* Qb    = (bf16_t*)walloc(M * EE * sizeof(bf16_t));          // 8 MB
  bf16_t* Kb    = (bf16_t*)walloc(M * EE * sizeof(bf16_t));          // 8 MB
  bf16_t* Vtb   = (bf16_t*)walloc(M * EE * sizeof(bf16_t));          // 8 MB
  bf16_t* valsb = (bf16_t*)walloc(M * EE * sizeof(bf16_t));          // 8 MB

  // 1) casts (everything then lives in the 192 MB L2 for the GEMMs)
  cast_f32_to_bf16<<<2048, 256, 0, stream>>>(x, xb, M * EE);
  cast_f32_to_bf16<<<2048, 256, 0, stream>>>(qkv_w, wqkvb, (size_t)3 * EE * EE);
  cast_f32_to_bf16<<<1024, 256, 0, stream>>>(o_w, owb, (size_t)EE * EE);

  // 2) QKV projection: [4096,1024] x [3072,1024]^T
  qkv_gemm_kernel<<<dim3(M / 128, (3 * EE) / 64), 256, 0, stream>>>(
      xb, wqkvb, qkv_b, Qb, Kb, Vtb);

  // 3) fused attention, streaming prev once
  attn_kernel<<<dim3(SS / 128, HH, BB), 256, 0, stream>>>(
      Qb, Kb, Vtb, prev, mask, valsb);

  // 4) output projection: [4096,1024] x [1024,1024]^T + bias -> f32
  oproj_gemm_kernel<<<dim3(M / 128, EE / 64), 256, 0, stream>>>(
      valsb, owb, o_b, out);
}
